// Fingerprint_87213605912981
// MI455X (gfx1250) — compile-verified, matrix operand order checked by reading, not run on hardware
//
#include <hip/hip_runtime.h>
#include <hip/hip_bf16.h>
#include <math.h>

// ---------------- problem constants (from reference setup_inputs) -----------
constexpr int cB  = 256;
constexpr int cL  = 128;
constexpr int cD  = 6;
constexpr int cF  = 200;
constexpr int cFA = 39;
constexpr int cFB = 10;
constexpr int cR  = 2;
constexpr int cNPF = (cF + 31) / 32;     // features per lane in wave32 = 7
constexpr int cKP  = 224;                // K padded to multiple of 32
constexpr int cNP1 = 256;                // Npad for N=200 (attend)
constexpr int cNP3 = 640;                // Npad for N=600 (GRU)

typedef __attribute__((ext_vector_type(16))) __bf16 bf16x16;
typedef __attribute__((ext_vector_type(8)))  float  floatx8;
typedef __attribute__((ext_vector_type(4)))  int    v4i;

#define GLOBAL_AS __attribute__((address_space(1)))
#define LDS_AS    __attribute__((address_space(3)))

#if __has_builtin(__builtin_amdgcn_global_load_async_to_lds_b128) && \
    __has_builtin(__builtin_amdgcn_s_wait_asynccnt)
#define USE_ASYNC_LDS 1
#endif

union FragU { uint4 q[2]; bf16x16 v; };

__device__ __forceinline__ unsigned short f2bf(float f) {
  // round-to-nearest-even fp32 -> bf16
  unsigned int u = __float_as_uint(f);
  u += 0x7fffu + ((u >> 16) & 1u);
  return (unsigned short)(u >> 16);
}

__device__ __forceinline__ float leaky(float x) { return x > 0.f ? x : 0.01f * x; }

__device__ __forceinline__ float wred(float v) {
  #pragma unroll
  for (int o = 16; o > 0; o >>= 1) v += __shfl_xor(v, o, 32);
  return v;
}

// ---------------------------------------------------------------------------
// Kernel 0a: zero the K-pad columns [cF, cKP) of the bf16 activation buffers.
// Producers only ever write f < cF, so pads stay zero for the whole launch.
// ---------------------------------------------------------------------------
__global__ __launch_bounds__(256) void pad_zero_kernel(
    unsigned short* __restrict__ b0, unsigned short* __restrict__ b1,
    unsigned short* __restrict__ b2)
{
  int idx = blockIdx.x * blockDim.x + threadIdx.x;
  const int PAD = cKP - cF;   // 24
  if (idx >= cB * cL * PAD) return;
  int m = idx / PAD;
  int c = idx - m * PAD;
  size_t o = (size_t)m * cKP + cF + c;
  b0[o] = 0; b1[o] = 0; b2[o] = 0;
}

// ---------------------------------------------------------------------------
// Kernel 0b: convert one [N,K=cF] fp32 weight matrix into a zero-padded
// [Npad, cKP] bf16 matrix (pads in both N and K directions).
// ---------------------------------------------------------------------------
__global__ __launch_bounds__(256) void convert_weights_kernel(
    const float* __restrict__ W, unsigned short* __restrict__ out,
    int N, int Npad)
{
  int idx = blockIdx.x * blockDim.x + threadIdx.x;
  if (idx >= Npad * cKP) return;
  int n = idx / cKP;
  int k = idx - n * cKP;
  out[idx] = (n < N && k < cF) ? f2bf(W[(size_t)n * cF + k]) : (unsigned short)0;
}

// ---------------------------------------------------------------------------
// Kernel 1: fused initial embeddings.
//   cur    = leaky_relu(atom_list @ atom_W.T + atom_b)  (fp32 + bf16 shadow)
//   A_emb  = atom_list @ nbr_W[:, :FA].T
//   B_emb  = bond_list @ nbr_W[:, FA:].T
// ---------------------------------------------------------------------------
__global__ __launch_bounds__(256) void init_embed_kernel(
    const float* __restrict__ atom_list, const float* __restrict__ bond_list,
    const float* __restrict__ atomW, const float* __restrict__ atomB,
    const float* __restrict__ nbrW,
    float* __restrict__ cur, unsigned short* __restrict__ cur_bf,
    float* __restrict__ A_emb, float* __restrict__ B_emb)
{
  int idx = blockIdx.x * blockDim.x + threadIdx.x;
  if (idx >= cB * cL * cF) return;
  int m = idx / cF;
  int f = idx - m * cF;
  const float* ar = atom_list + (size_t)m * cFA;
  const float* br = bond_list + (size_t)m * cFB;
  const float* w1 = atomW + (size_t)f * cFA;
  const float* w2 = nbrW  + (size_t)f * (cFA + cFB);
  float a1 = atomB[f], a2 = 0.f, a3 = 0.f;
  #pragma unroll
  for (int k = 0; k < cFA; ++k) { float x = ar[k]; a1 = fmaf(x, w1[k], a1); a2 = fmaf(x, w2[k], a2); }
  #pragma unroll
  for (int k = 0; k < cFB; ++k) a3 = fmaf(br[k], w2[cFA + k], a3);
  float h = leaky(a1);
  cur[idx] = h;
  cur_bf[(size_t)m * cKP + f] = f2bf(h);
  A_emb[idx] = a2;
  B_emb[idx] = a3;
}

// ---------------------------------------------------------------------------
// Kernel 2: attention (one wave32 per atom; 8 waves per block).
// agg_bf[i,:] = bf16( sum_j attn_ij * nbr_ij ),  ssum[i] = sum_j attn_ij.
// Round 0 neighbor rows: leaky(A_emb[a]+B_emb[b]+nbr_b); later rounds gather
// rows of fp32 cur (L2-resident: 100KB/molecule vs 192MB L2).
// ---------------------------------------------------------------------------
__global__ __launch_bounds__(256) void attention_kernel(
    const float* __restrict__ cur,
    const float* __restrict__ A_emb, const float* __restrict__ B_emb,
    const float* __restrict__ nbr_b,
    const int*   __restrict__ atom_deg, const int* __restrict__ bond_deg,
    const float* __restrict__ alignW,   // [2F] for this round
    const float* __restrict__ alignB,   // align_b base (R,1)
    int d, int round0,
    unsigned short* __restrict__ agg_bf, float* __restrict__ ssum)
{
  const int lane = threadIdx.x & 31;
  const int atom = blockIdx.x * 8 + (threadIdx.x >> 5);
  if (atom >= cB * cL) return;
  const int b = atom / cL;
  const float ab = alignB[d];

  // q_i = cur_i . wq
  const float* crow = cur + (size_t)atom * cF;
  float qp = 0.f;
  for (int f = lane; f < cF; f += 32) qp = fmaf(crow[f], alignW[f], qp);
  const float q = wred(qp);

  float nv[cD][cNPF];
  int   ai[cD];
  float score[cD];
  #pragma unroll
  for (int j = 0; j < cD; ++j) {
    const int a  = atom_deg[(size_t)atom * cD + j];
    const int bd = bond_deg[(size_t)atom * cD + j];
    ai[j] = a;
    const size_t nrow = (size_t)(b * cL + a)  * cF;
    const size_t brow = (size_t)(b * cL + bd) * cF;
    float kp = 0.f;
    int c = 0;
    for (int f = lane; f < cF; f += 32, ++c) {
      float v;
      if (round0) {
        v = leaky(A_emb[nrow + f] + B_emb[brow + f] + nbr_b[f]);
      } else {
        v = cur[nrow + f];
      }
      nv[j][c] = v;
      kp = fmaf(v, alignW[cF + f], kp);
    }
    float sc = leaky(q + wred(kp) + ab);
    if (a == cL - 1) sc += -9e8f;   // softmax_mask on padding slots
    score[j] = sc;
  }

  // softmax over D (wave-uniform after reductions)
  float mx = score[0];
  #pragma unroll
  for (int j = 1; j < cD; ++j) mx = fmaxf(mx, score[j]);
  float ex[cD], den = 0.f;
  #pragma unroll
  for (int j = 0; j < cD; ++j) { ex[j] = expf(score[j] - mx); den += ex[j]; }
  const float inv = 1.f / den;

  float av[cNPF];
  #pragma unroll
  for (int c = 0; c < cNPF; ++c) av[c] = 0.f;
  float s = 0.f;
  #pragma unroll
  for (int j = 0; j < cD; ++j) {
    float a_ = ex[j] * inv;
    if (ai[j] == cL - 1) a_ = 0.f;  // attend_mask
    s += a_;
    #pragma unroll
    for (int c = 0; c < cNPF; ++c) av[c] = fmaf(a_, nv[j][c], av[c]);
  }
  int c = 0;
  for (int f = lane; f < cF; f += 32, ++c)
    agg_bf[(size_t)atom * cKP + f] = f2bf(av[c]);
  if (lane == 0) ssum[atom] = s;
}

// ---------------------------------------------------------------------------
// Kernel 3: bf16 WMMA GEMM  OUT[M,N] = epilogue( A[M,K] @ W[N,K]^T )
// A: bf16 [M, cKP] (K-pads zero).  W: bf16 [Npad, cKP] (fully zero-padded).
//   MODE 0: out_bf16[M,cKP] = bf16(elu(acc + svec[m]*bias[n]))  (context)
//   MODE 1: out_f32 [M,N]   = acc + bias[n]                     (GRU proj)
// Block: 256 threads = 8 waves; BM=128, BN=64, BK=32; 7 unrolled k-chunks.
// Staging uses GLOBAL_LOAD_ASYNC_TO_LDS_B128 when the toolchain exposes it
// (CDNA5 async DMA: no VGPR round-trip, tracked by ASYNCcnt, overlapped with
// WMMA on the other LDS buffer); falls back to uint4 register copies.
// ---------------------------------------------------------------------------
template<int MODE>
__global__ __launch_bounds__(256) void wmma_gemm_kernel(
    const unsigned short* __restrict__ A, const unsigned short* __restrict__ W,
    const float* __restrict__ bias, const float* __restrict__ svec,
    void* __restrict__ outv, int M, int N)
{
  constexpr int NCH = cKP / 32;   // 7
  __shared__ __align__(16) unsigned short As[2][128 * 32];
  __shared__ __align__(16) unsigned short Ws[2][64 * 32];
  const int tid  = threadIdx.x;
  const int lane = tid & 31;
  const int wave = tid >> 5;
  const int half = lane >> 4;
  const int l16  = lane & 15;
  const int m0 = blockIdx.x * 128;
  const int n0 = blockIdx.y * 64;

  const floatx8 z8 = {0.f, 0.f, 0.f, 0.f, 0.f, 0.f, 0.f, 0.f};
  floatx8 acc[4];
  #pragma unroll
  for (int t = 0; t < 4; ++t) acc[t] = z8;

  const int arow = tid >> 1;           // 0..127
  const int acb  = (tid & 1) * 16;     // 0 / 16
  const int wrow = tid >> 2;           // 0..63
  const int wcb  = (tid & 3) * 8;      // 0..24
  const unsigned short* aPtr = A + (size_t)(m0 + arow) * cKP + acb;
  const unsigned short* wPtr = W + (size_t)(n0 + wrow) * cKP + wcb;

#if defined(USE_ASYNC_LDS)
  // -------- CDNA5 async global->LDS DMA staging (ASYNCcnt-tracked) --------
  auto stage = [&](int buf, int k0) {
    __builtin_amdgcn_global_load_async_to_lds_b128(
        (GLOBAL_AS v4i*)(aPtr + k0),
        (LDS_AS v4i*)&As[buf][arow * 32 + acb], 0, 0);
    __builtin_amdgcn_global_load_async_to_lds_b128(
        (GLOBAL_AS v4i*)(aPtr + k0 + 8),
        (LDS_AS v4i*)&As[buf][arow * 32 + acb + 8], 0, 0);
    __builtin_amdgcn_global_load_async_to_lds_b128(
        (GLOBAL_AS v4i*)(wPtr + k0),
        (LDS_AS v4i*)&Ws[buf][wrow * 32 + wcb], 0, 0);
  };
  auto stageWait = [&]() { __builtin_amdgcn_s_wait_asynccnt(0); };
#else
  // ---------------- fallback: registers round-trip staging ----------------
  uint4 aReg0, aReg1, wReg;
  auto stage = [&](int buf, int k0) {
    aReg0 = *reinterpret_cast<const uint4*>(aPtr + k0);
    aReg1 = *reinterpret_cast<const uint4*>(aPtr + k0 + 8);
    wReg  = *reinterpret_cast<const uint4*>(wPtr + k0);
    uint4* ad = reinterpret_cast<uint4*>(&As[buf][arow * 32 + acb]);
    ad[0] = aReg0;
    ad[1] = aReg1;
    *reinterpret_cast<uint4*>(&Ws[buf][wrow * 32 + wcb]) = wReg;
  };
  auto stageWait = [&]() {};
#endif

  auto mmaStep = [&](int buf) {
    FragU af;
    const unsigned short* ap = &As[buf][(wave * 16 + l16) * 32];
    af.q[0] = *reinterpret_cast<const uint4*>(ap + half * 8);
    af.q[1] = *reinterpret_cast<const uint4*>(ap + 16 + half * 8);
    #pragma unroll
    for (int t = 0; t < 4; ++t) {
      FragU bfv;
      const unsigned short* bp = &Ws[buf][(t * 16 + l16) * 32];
      bfv.q[0] = *reinterpret_cast<const uint4*>(bp + half * 8);
      bfv.q[1] = *reinterpret_cast<const uint4*>(bp + 16 + half * 8);
      acc[t] = __builtin_amdgcn_wmma_f32_16x16x32_bf16(
          false, af.v, false, bfv.v, (short)0, acc[t], false, false);
    }
  };

  stage(0, 0);
  stageWait();
  __syncthreads();
  #pragma unroll
  for (int kc = 0; kc < NCH; ++kc) {
    if (kc + 1 < NCH) stage((kc + 1) & 1, (kc + 1) * 32);  // overlap with WMMA
    mmaStep(kc & 1);
    if (kc + 1 < NCH) {
      stageWait();
      __syncthreads();
    }
  }

  // epilogue: C/D layout -> VGPR r: row = r + (lane>=16)*8, col = lane&15
  const int col    = l16;
  const int rowOff = half * 8;
  #pragma unroll
  for (int t = 0; t < 4; ++t) {
    const int gn = n0 + t * 16 + col;
    if (gn >= N) continue;
    #pragma unroll
    for (int r = 0; r < 8; ++r) {
      const int gm = m0 + wave * 16 + rowOff + r;
      float v = acc[t][r];
      if (MODE == 0) {
        v += svec[gm] * bias[gn];
        v = (v > 0.f) ? v : (expf(v) - 1.f);   // elu
        ((unsigned short*)outv)[(size_t)gm * cKP + gn] = f2bf(v);
      } else {
        v += bias[gn];
        ((float*)outv)[(size_t)gm * N + gn] = v;
      }
    }
  }
}

// ---------------------------------------------------------------------------
// Kernel 4: GRU gates + relu.  cur_new = relu((1-z)*n + z*cur)
// Writes fp32 result (cur or d_out) and the bf16 shadow for the next GEMM.
// ---------------------------------------------------------------------------
__global__ __launch_bounds__(256) void gru_gate_kernel(
    const float* __restrict__ gi, const float* __restrict__ gh,
    const float* __restrict__ cur, float* __restrict__ out,
    unsigned short* __restrict__ cur_bf)
{
  int idx = blockIdx.x * blockDim.x + threadIdx.x;
  if (idx >= cB * cL * cF) return;
  int m = idx / cF;
  int f = idx - m * cF;
  size_t base = (size_t)m * 3 * cF;
  float ir = gi[base + f], iz = gi[base + cF + f], in = gi[base + 2 * cF + f];
  float hr = gh[base + f], hz = gh[base + cF + f], hn = gh[base + 2 * cF + f];
  float r = 1.f / (1.f + expf(-(ir + hr)));
  float z = 1.f / (1.f + expf(-(iz + hz)));
  float n = tanhf(in + r * hn);
  float h = (1.f - z) * n + z * cur[idx];
  float o = fmaxf(h, 0.f);
  out[idx] = o;
  cur_bf[(size_t)m * cKP + f] = f2bf(o);
}

// ---------------------------------------------------------------------------
extern "C" void kernel_launch(void* const* d_in, const int* in_sizes, int n_in,
                              void* d_out, int out_size, void* d_ws, size_t ws_size,
                              hipStream_t stream) {
  const float* atom_list   = (const float*)d_in[0];
  const float* bond_list   = (const float*)d_in[1];
  const int*   atom_degree = (const int*)  d_in[2];
  const int*   bond_degree = (const int*)  d_in[3];
  const float* atom_W      = (const float*)d_in[4];
  const float* atom_b      = (const float*)d_in[5];
  const float* nbr_W       = (const float*)d_in[6];
  const float* nbr_b       = (const float*)d_in[7];
  const float* align_W     = (const float*)d_in[8];
  const float* align_b     = (const float*)d_in[9];
  const float* attend_W    = (const float*)d_in[10];
  const float* attend_b    = (const float*)d_in[11];
  const float* gru_Wih     = (const float*)d_in[12];
  const float* gru_Whh     = (const float*)d_in[13];
  const float* gru_bih     = (const float*)d_in[14];
  const float* gru_bhh     = (const float*)d_in[15];

  const size_t BLF = (size_t)cB * cL * cF;      // 6,553,600
  const int    M   = cB * cL;                   // 32768
  const size_t MKP = (size_t)M * cKP;           // bf16 activation row-padded

  float* ws    = (float*)d_ws;
  float* cur   = ws;
  float* A_emb = ws + BLF;
  float* B_emb = ws + 2 * BLF;
  float* gi    = ws + 3 * BLF;                  // 3*BLF
  float* gh    = ws + 6 * BLF;                  // 3*BLF
  float* ssum  = ws + 9 * BLF;                  // M
  unsigned short* bf_base    = (unsigned short*)(ws + 9 * BLF + M);
  unsigned short* cur_bf     = bf_base;
  unsigned short* agg_bf     = bf_base + MKP;
  unsigned short* context_bf = bf_base + 2 * MKP;
  unsigned short* attW_bf    = bf_base + 3 * MKP;                 // R * 256*224
  unsigned short* wih_bf     = attW_bf + (size_t)cR * cNP1 * cKP; // R * 640*224
  unsigned short* whh_bf     = wih_bf  + (size_t)cR * cNP3 * cKP;

  const int total = (int)BLF;
  const int eb = (total + 255) / 256;

  // --- one-time per launch: zero K-pads + convert weights to padded bf16 ---
  {
    int padTot = M * (cKP - cF);
    pad_zero_kernel<<<(padTot + 255) / 256, 256, 0, stream>>>(
        cur_bf, agg_bf, context_bf);
    int w1 = cNP1 * cKP, w3 = cNP3 * cKP;
    for (int d = 0; d < cR; ++d) {
      convert_weights_kernel<<<(w1 + 255) / 256, 256, 0, stream>>>(
          attend_W + (size_t)d * cF * cF, attW_bf + (size_t)d * w1, cF, cNP1);
      convert_weights_kernel<<<(w3 + 255) / 256, 256, 0, stream>>>(
          gru_Wih + (size_t)d * 3 * cF * cF, wih_bf + (size_t)d * w3, 3 * cF, cNP3);
      convert_weights_kernel<<<(w3 + 255) / 256, 256, 0, stream>>>(
          gru_Whh + (size_t)d * 3 * cF * cF, whh_bf + (size_t)d * w3, 3 * cF, cNP3);
    }
  }

  init_embed_kernel<<<eb, 256, 0, stream>>>(
      atom_list, bond_list, atom_W, atom_b, nbr_W, cur, cur_bf, A_emb, B_emb);

  for (int d = 0; d < cR; ++d) {
    attention_kernel<<<M / 8, 256, 0, stream>>>(
        cur, A_emb, B_emb, nbr_b, atom_degree, bond_degree,
        align_W + (size_t)d * 2 * cF, align_b, d, (d == 0) ? 1 : 0, agg_bf, ssum);

    dim3 gc(M / 128, cNP1 / 64);   // 256 x 4
    wmma_gemm_kernel<0><<<gc, 256, 0, stream>>>(
        agg_bf, attW_bf + (size_t)d * cNP1 * cKP, attend_b + (size_t)d * cF,
        ssum, context_bf, M, cF);

    dim3 gg(M / 128, cNP3 / 64);   // 256 x 10
    wmma_gemm_kernel<1><<<gg, 256, 0, stream>>>(
        context_bf, wih_bf + (size_t)d * cNP3 * cKP, gru_bih + (size_t)d * 3 * cF,
        nullptr, gi, M, 3 * cF);
    wmma_gemm_kernel<1><<<gg, 256, 0, stream>>>(
        cur_bf, whh_bf + (size_t)d * cNP3 * cKP, gru_bhh + (size_t)d * 3 * cF,
        nullptr, gh, M, 3 * cF);

    float* outp = (d == cR - 1) ? (float*)d_out : cur;
    gru_gate_kernel<<<eb, 256, 0, stream>>>(gi, gh, cur, outp, cur_bf);
  }
}
